// GenomeModel_20461224198259
// MI455X (gfx1250) — matrix operand order repaired and checked by reading, not compiled, not run on hardware
//
#include <hip/hip_runtime.h>
#include <hip/hip_bf16.h>
#include <math.h>

typedef __attribute__((ext_vector_type(16))) _Float16 v16h;
typedef __attribute__((ext_vector_type(8)))  _Float16 v8h;
typedef __attribute__((ext_vector_type(8)))  float    v8f;

#define DIM_D 512
#define DIM_F 2048
#define DIM_V 32000
#define NTOK  4096          // B*S = 4*1024

#define BM 128              // block tile M
#define BN 128              // block tile N
#define BK 32               // K step (= one WMMA K)
#define ASTR (BK + 8)       // 40 halfs (80B row): 16B-aligned frags, staggered banks
#define BSTR (BK + 8)

// MODE 0: A = gather(embed, tokens) fp32->f16, B = w1, epilogue = SiLU, out = h (f16)
// MODE 1: A = h (f16),               B = w4, epilogue = + embed[tok], out = r_pre (f32)
// MODE 2: A = r (f16),               B = w8, epilogue = none,         out = logits (f32)
template<int MODE>
__global__ __launch_bounds__(256)
void wmma_gemm(const float* __restrict__ embed,
               const _Float16* __restrict__ Ah,
               const float* __restrict__ Bw,
               const int* __restrict__ tokens,
               float* __restrict__ outF,
               _Float16* __restrict__ outH,
               int K, int N)
{
    __shared__ _Float16 As[2][BM * ASTR];   // 2 x 10 KB
    __shared__ _Float16 Bs[2][BN * BSTR];   // 2 x 10 KB  (stored transposed: [n][k])
    __shared__ int      tok[BM];

    const int t     = threadIdx.x;
    const int mBase = blockIdx.y * BM;
    const int nBase = blockIdx.x * BN;

    if (MODE != 2) {
        if (t < BM) tok[t] = tokens[mBase + t];
        __syncthreads();                    // tok[] visible for MODE0 gather fill
    }

    // A fill: 128 rows x 32 k, 16 halfs/thread (two 16B LDS stores)
    const int ar = t >> 1;                  // 0..127
    const int ac = (t & 1) << 4;            // 0 or 16
    // B fill: 32 k-rows x 128 n, 16 fp32/thread read contiguous along N
    const int bk = t >> 3;                  // 0..31
    const int bn = (t & 7) << 4;            // 0..112

    const int wave = t >> 5;                // 8 wave32 waves
    const int lane = t & 31;
    const int wsm  = wave & 3;              // M strip of 32 (2 subtiles)
    const int wsn  = wave >> 2;             // N strip of 64 (4 subtiles)
    const int lhi  = lane >> 4;
    const int llo  = lane & 15;

    auto fillA = [&](int buf, int kb) {
        if (MODE == 0) {
            const float* src = embed + (size_t)tok[ar] * DIM_D + kb + ac;
            _Float16 tmp[16];
            #pragma unroll
            for (int i = 0; i < 16; ++i) tmp[i] = (_Float16)src[i];
            *(v8h*)&As[buf][ar * ASTR + ac]     = *(const v8h*)&tmp[0];
            *(v8h*)&As[buf][ar * ASTR + ac + 8] = *(const v8h*)&tmp[8];
        } else {
            const _Float16* src = Ah + (size_t)(mBase + ar) * K + kb + ac;
            *(v8h*)&As[buf][ar * ASTR + ac]     = *(const v8h*)(src);
            *(v8h*)&As[buf][ar * ASTR + ac + 8] = *(const v8h*)(src + 8);
        }
    };
    auto fillB = [&](int buf, int kb) {
        const float* src = Bw + (size_t)(kb + bk) * N + nBase + bn;
        #pragma unroll
        for (int i = 0; i < 16; ++i)
            Bs[buf][(bn + i) * BSTR + bk] = (_Float16)src[i];
    };

    v8f acc[2][4] = {};

    fillA(0, 0);
    fillB(0, 0);

    const int nIter = K / BK;
    for (int it = 0; it < nIter; ++it) {
        const int cur = it & 1;
        __syncthreads();                    // fill(cur) done; prev compute on cur^1 done
        if (it + 1 < nIter) {               // overlap next fill with this step's WMMAs
            fillA(cur ^ 1, (it + 1) * BK);
            fillB(cur ^ 1, (it + 1) * BK);
        }

        // A fragments: 16-bit 16x32 layout, k = {lhi*8..+7} and {16+lhi*8..+7}
        v16h a[2];
        #pragma unroll
        for (int im = 0; im < 2; ++im) {
            const int m = wsm * 32 + im * 16 + llo;
            v8h alo = *(const v8h*)&As[cur][m * ASTR + lhi * 8];
            v8h ahi = *(const v8h*)&As[cur][m * ASTR + 16 + lhi * 8];
            #pragma unroll
            for (int i = 0; i < 8; ++i) { a[im][i] = alo[i]; a[im][8 + i] = ahi[i]; }
        }
        // B fragments: k = {lhi*8..+7} and {16+lhi*8..+7}, matching A
        v16h b[4];
        #pragma unroll
        for (int jn = 0; jn < 4; ++jn) {
            const int n = wsn * 64 + jn * 16 + llo;
            v8h blo = *(const v8h*)&Bs[cur][n * BSTR + lhi * 8];
            v8h bhi = *(const v8h*)&Bs[cur][n * BSTR + 16 + lhi * 8];
            #pragma unroll
            for (int i = 0; i < 8; ++i) { b[jn][i] = blo[i]; b[jn][8 + i] = bhi[i]; }
        }
        // 8 WMMAs per wave per K-step
        #pragma unroll
        for (int im = 0; im < 2; ++im)
            #pragma unroll
            for (int jn = 0; jn < 4; ++jn)
                acc[im][jn] = __builtin_amdgcn_wmma_f32_16x16x32_f16(
                                  false, a[im], false, b[jn], (short)0,
                                  acc[im][jn], false, false);
    }

    // Epilogue: C layout VGPR i -> m = +lhi*8 + i, n = llo
    #pragma unroll
    for (int im = 0; im < 2; ++im) {
        #pragma unroll
        for (int jn = 0; jn < 4; ++jn) {
            const int col = nBase + wsn * 64 + jn * 16 + llo;
            #pragma unroll
            for (int i = 0; i < 8; ++i) {
                const int rloc = wsm * 32 + im * 16 + lhi * 8 + i;
                const int row  = mBase + rloc;
                float v = acc[im][jn][i];
                if (MODE == 0) {            // SiLU via fast v_rcp_f32
                    float s = v * __builtin_amdgcn_rcpf(1.0f + __expf(-v));
                    outH[(size_t)row * DIM_F + col] = (_Float16)s;
                } else if (MODE == 1) {     // residual add (re-gather embedding)
                    float e = embed[(size_t)tok[rloc] * DIM_D + col];
                    outF[(size_t)row * DIM_D + col] = v + e;
                } else {
                    outF[(size_t)row * (size_t)DIM_V + col] = v;
                }
            }
        }
    }
}

// Row-wise L2 norm: r = r_pre / (||r_pre||_2 + eps), emitted as f16 for GEMM3's A.
__global__ __launch_bounds__(256)
void l2norm_kernel(const float* __restrict__ rp, _Float16* __restrict__ r)
{
    const int row = blockIdx.x;
    const float* p = rp + (size_t)row * DIM_D;

    float s = 0.0f;
    for (int i = threadIdx.x; i < DIM_D; i += 256) { float v = p[i]; s += v * v; }
    #pragma unroll
    for (int off = 16; off; off >>= 1) s += __shfl_xor(s, off, 32);

    __shared__ float red[8];
    __shared__ float inv;
    if ((threadIdx.x & 31) == 0) red[threadIdx.x >> 5] = s;
    __syncthreads();
    if (threadIdx.x == 0) {
        float tot = 0.0f;
        #pragma unroll
        for (int i = 0; i < 8; ++i) tot += red[i];
        inv = 1.0f / (sqrtf(tot) + 1e-8f);
    }
    __syncthreads();
    const float sc = inv;
    for (int i = threadIdx.x; i < DIM_D; i += 256)
        r[(size_t)row * DIM_D + i] = (_Float16)(p[i] * sc);
}

extern "C" void kernel_launch(void* const* d_in, const int* in_sizes, int n_in,
                              void* d_out, int out_size, void* d_ws, size_t ws_size,
                              hipStream_t stream)
{
    const int*   x     = (const int*)d_in[0];     // (B,S) token ids
    const float* embed = (const float*)d_in[1];   // (V,D)
    const float* w1    = (const float*)d_in[2];   // (D,F)
    const float* w4    = (const float*)d_in[3];   // (F,D)
    const float* w8    = (const float*)d_in[4];   // (D,V)
    float* out = (float*)d_out;                   // (B,S,V) fp32 logits

    // Workspace layout (~28.4 MB total)
    char* ws = (char*)d_ws;
    _Float16* h  = (_Float16*)ws;                                   // 4096x2048 f16 = 16 MB
    float*    rp = (float*)(ws + (size_t)NTOK * DIM_F * 2);         // 4096x512  f32 =  8 MB
    _Float16* r  = (_Float16*)(ws + (size_t)NTOK * DIM_F * 2
                                  + (size_t)NTOK * DIM_D * 4);      // 4096x512  f16 =  4 MB

    dim3 blk(256);
    // h = silu(embed[x] @ w1)
    wmma_gemm<0><<<dim3(DIM_F / BN, NTOK / BM), blk, 0, stream>>>(
        embed, nullptr, w1, x, nullptr, h, DIM_D, DIM_F);
    // r_pre = embed[x] + h @ w4
    wmma_gemm<1><<<dim3(DIM_D / BN, NTOK / BM), blk, 0, stream>>>(
        embed, h, w4, x, rp, nullptr, DIM_F, DIM_D);
    // r = l2norm(r_pre)
    l2norm_kernel<<<dim3(NTOK), blk, 0, stream>>>(rp, r);
    // logits = r @ w8
    wmma_gemm<2><<<dim3(DIM_V / BN, NTOK / BM), blk, 0, stream>>>(
        nullptr, r, w8, nullptr, out, nullptr, DIM_D, DIM_V);
}
